// HybridQuantumKernelModel_65481071407188
// MI455X (gfx1250) — matrix operand order repaired and checked by reading, not compiled, not run on hardware
//
#include <hip/hip_runtime.h>
#include <hip/hip_bf16.h>

typedef _Float16 v8h  __attribute__((ext_vector_type(8)));
typedef _Float16 v16h __attribute__((ext_vector_type(16)));
typedef float    v8f  __attribute__((ext_vector_type(8)));

#define NB    4096      // batch size for both x and y
#define KDIM  784       // 28*28 flattened pixels
#define KPAD  800       // 25 chunks of 32 (zero padded)
#define NCHUNK 25

// ---------------------------------------------------------------------------
// Convert x rows -> row-major padded f16, and compute ||x_i||^2 (f32, exact).
// ---------------------------------------------------------------------------
__global__ __launch_bounds__(256) void convert_rows_kernel(
    const float* __restrict__ src, _Float16* __restrict__ dst,
    float* __restrict__ norms)
{
    __shared__ float s_red[256];
    const int tid = threadIdx.x;
    const int row = blockIdx.x;
    const float* s = src + (size_t)row * KDIM;
    _Float16* d   = dst + (size_t)row * KPAD;
    float ps = 0.f;
    for (int k = tid; k < KPAD; k += 256) {
        float v = (k < KDIM) ? s[k] : 0.f;
        d[k] = (_Float16)v;
        ps += v * v;
    }
    s_red[tid] = ps;
    __syncthreads();
    for (int off = 128; off > 0; off >>= 1) {
        if (tid < off) s_red[tid] += s_red[tid + off];
        __syncthreads();
    }
    if (tid == 0) norms[row] = s_red[0];
}

// ---------------------------------------------------------------------------
// Convert y rows -> K-major (transposed) f16 so B-fragments load contiguously,
// and compute ||y_j||^2.  dstT is KPAD x NB.
// ---------------------------------------------------------------------------
__global__ __launch_bounds__(256) void convert_cols_kernel(
    const float* __restrict__ src, _Float16* __restrict__ dstT,
    float* __restrict__ norms)
{
    __shared__ float s_red[256];
    const int tid = threadIdx.x;
    const int row = blockIdx.x;             // image index j
    const float* s = src + (size_t)row * KDIM;
    float ps = 0.f;
    for (int k = tid; k < KPAD; k += 256) {
        float v = (k < KDIM) ? s[k] : 0.f;
        dstT[(size_t)k * NB + row] = (_Float16)v;
        ps += v * v;
    }
    s_red[tid] = ps;
    __syncthreads();
    for (int off = 128; off > 0; off >>= 1) {
        if (tid < off) s_red[tid] += s_red[tid + off];
        __syncthreads();
    }
    if (tid == 0) norms[row] = s_red[0];
}

// ---------------------------------------------------------------------------
// Per-image CNN embedding (pre-BatchNorm): conv1->relu->pool -> conv2->relu->
// pool -> fc1->relu -> fc2.  One block per image; all activations in LDS.
// ---------------------------------------------------------------------------
__global__ __launch_bounds__(256) void embed_kernel(
    const float* __restrict__ img,
    const float* __restrict__ c1w, const float* __restrict__ c1b,
    const float* __restrict__ c2w, const float* __restrict__ c2b,
    const float* __restrict__ f1w, const float* __restrict__ f1b,
    const float* __restrict__ f2w, const float* __restrict__ f2b,
    float* __restrict__ emb_raw)
{
    __shared__ float s_img[784];      // 1x28x28
    __shared__ float s_c1[6272];      // 8x28x28
    __shared__ float s_p1[1568];      // 8x14x14
    __shared__ float s_c2[3136];      // 16x14x14
    __shared__ float s_p2[784];       // 16x7x7
    __shared__ float s_h[64];
    const int tid = threadIdx.x;
    const int b   = blockIdx.x;
    const float* im = img + (size_t)b * 784;
    for (int i = tid; i < 784; i += 256) s_img[i] = im[i];
    __syncthreads();
    // conv1 (SAME, 3x3) + relu
    for (int idx = tid; idx < 6272; idx += 256) {
        int oc = idx / 784, p = idx % 784, oy = p / 28, ox = p % 28;
        float acc = c1b[oc];
        for (int ky = 0; ky < 3; ++ky) {
            int iy = oy + ky - 1;
            if (iy < 0 || iy >= 28) continue;
            for (int kx = 0; kx < 3; ++kx) {
                int ix = ox + kx - 1;
                if (ix < 0 || ix >= 28) continue;
                acc += s_img[iy * 28 + ix] * c1w[oc * 9 + ky * 3 + kx];
            }
        }
        s_c1[idx] = fmaxf(acc, 0.f);
    }
    __syncthreads();
    // maxpool 2x2 -> 8x14x14
    for (int idx = tid; idx < 1568; idx += 256) {
        int c = idx / 196, p = idx % 196, y = p / 14, x = p % 14;
        const float* base = s_c1 + c * 784 + (y * 2) * 28 + x * 2;
        s_p1[idx] = fmaxf(fmaxf(base[0], base[1]), fmaxf(base[28], base[29]));
    }
    __syncthreads();
    // conv2 (SAME, 3x3, 8->16) + relu
    for (int idx = tid; idx < 3136; idx += 256) {
        int oc = idx / 196, p = idx % 196, oy = p / 14, ox = p % 14;
        float acc = c2b[oc];
        for (int ic = 0; ic < 8; ++ic) {
            const float* w   = c2w + (oc * 8 + ic) * 9;
            const float* inp = s_p1 + ic * 196;
            for (int ky = 0; ky < 3; ++ky) {
                int iy = oy + ky - 1;
                if (iy < 0 || iy >= 14) continue;
                for (int kx = 0; kx < 3; ++kx) {
                    int ix = ox + kx - 1;
                    if (ix < 0 || ix >= 14) continue;
                    acc += inp[iy * 14 + ix] * w[ky * 3 + kx];
                }
            }
        }
        s_c2[idx] = fmaxf(acc, 0.f);
    }
    __syncthreads();
    // maxpool 2x2 -> 16x7x7
    for (int idx = tid; idx < 784; idx += 256) {
        int c = idx / 49, p = idx % 49, y = p / 7, x = p % 7;
        const float* base = s_c2 + c * 196 + (y * 2) * 14 + x * 2;
        s_p2[idx] = fmaxf(fmaxf(base[0], base[1]), fmaxf(base[14], base[15]));
    }
    __syncthreads();
    // fc1 (784 -> 64) + relu
    if (tid < 64) {
        const float* w = f1w + (size_t)tid * 784;
        float acc = f1b[tid];
        for (int k = 0; k < 784; ++k) acc += s_p2[k] * w[k];
        s_h[tid] = fmaxf(acc, 0.f);
    }
    __syncthreads();
    // fc2 (64 -> 4)
    if (tid < 4) {
        const float* w = f2w + tid * 64;
        float acc = f2b[tid];
        for (int k = 0; k < 64; ++k) acc += s_h[k] * w[k];
        emb_raw[b * 4 + tid] = acc;
    }
}

// ---------------------------------------------------------------------------
// BatchNorm1d in training mode: per-batch biased stats over 4096 rows x 4 ch.
// One block; two-phase (stats then apply).
// ---------------------------------------------------------------------------
__global__ __launch_bounds__(256) void bn_kernel(
    const float* __restrict__ raw, const float* __restrict__ g,
    const float* __restrict__ bb, float* __restrict__ outp)
{
    __shared__ float s_sum[256], s_sq[256];
    __shared__ float s_scale[4], s_shift[4];
    const int tid = threadIdx.x;
    const int c = tid & 3, grp = tid >> 2;
    float ps = 0.f, pq = 0.f;
    for (int i = grp; i < NB; i += 64) {
        float v = raw[i * 4 + c];
        ps += v;
        pq += v * v;
    }
    s_sum[tid] = ps;
    s_sq[tid]  = pq;
    __syncthreads();
    if (tid < 4) {
        float su = 0.f, sq = 0.f;
        for (int gg = 0; gg < 64; ++gg) {
            su += s_sum[gg * 4 + tid];
            sq += s_sq[gg * 4 + tid];
        }
        float mean = su * (1.f / NB);
        float var  = sq * (1.f / NB) - mean * mean;   // biased variance
        float sc   = rsqrtf(var + 1e-5f) * g[tid];
        s_scale[tid] = sc;
        s_shift[tid] = bb[tid] - mean * sc;
    }
    __syncthreads();
    for (int idx = tid; idx < NB * 4; idx += 256) {
        int cc = idx & 3;
        outp[idx] = raw[idx] * s_scale[cc] + s_shift[cc];
    }
}

// ---------------------------------------------------------------------------
// Fused WMMA GEMM + RBF + quantum-cos epilogue, 2x4 register blocking.
// Block = 256 threads = 8 wave32s arranged 4(M) x 2(N); each wave computes a
// 32x64 tile (2 M-tiles x 4 N-tiles of 16x16) -> block tile 128x128.
// Per 32-K chunk: 2 A-frags + 4 B-frags (4 x b128 + 64 f16 loads) feed 8 WMMAs
// (0.75 frags/WMMA vs 2 unblocked -> ~2.7x less L2 fragment traffic).
// A frag (16x32 f16): lane l holds row M=l&15, K-half (l>>4)*8.
// B frag (32x16 f16): lane l holds col N=l&15, K as in A; the 4 N-tiles are
// 16 yhT columns apart; each element is one f16 from a yhT row (stride NB).
// ---------------------------------------------------------------------------
__global__ __launch_bounds__(256) void fused_rbf_wmma_kernel(
    const _Float16* __restrict__ xh, const _Float16* __restrict__ yhT,
    const float* __restrict__ nx, const float* __restrict__ ny,
    const float* __restrict__ ex, const float* __restrict__ ey,
    float* __restrict__ out)
{
    const int wave = threadIdx.x >> 5;
    const int lane = threadIdx.x & 31;
    const int row0 = blockIdx.x * 128 + (wave >> 1) * 32;   // wave M-tile base
    const int col0 = blockIdx.y * 128 + (wave & 1) * 64;    // wave N-tile base

    const int arow  = row0 + (lane & 15);
    const int khalf = (lane >> 4) * 8;
    const _Float16* ap = xh + (size_t)arow * KPAD + khalf;   // 16B aligned
    const _Float16* bp = yhT + (size_t)khalf * NB + col0 + (lane & 15);

    v8f acc[2][4] = {};
    for (int kc = 0; kc < NCHUNK; ++kc) {
        v16h a[2], b[4];
        #pragma unroll
        for (int mt = 0; mt < 2; ++mt) {
            const _Float16* am = ap + (size_t)mt * 16 * KPAD;
            v8h lo = *(const v8h*)am;
            v8h hi = *(const v8h*)(am + 16);
            #pragma unroll
            for (int e = 0; e < 8; ++e) { a[mt][e] = lo[e]; a[mt][e + 8] = hi[e]; }
        }
        #pragma unroll
        for (int nt = 0; nt < 4; ++nt) {
            const _Float16* bn = bp + nt * 16;
            #pragma unroll
            for (int e = 0; e < 8; ++e) { b[nt][e] = bn[(size_t)e * NB]; b[nt][e + 8] = bn[(size_t)(16 + e) * NB]; }
        }
        // prefetch next K chunk (both A m-tile rows + next B row)
        __builtin_prefetch(ap + 32, 0, 1);                       // global_prefetch_b8
        __builtin_prefetch(ap + (size_t)16 * KPAD + 32, 0, 1);
        __builtin_prefetch(bp + (size_t)32 * NB, 0, 1);
        __builtin_prefetch(bp + (size_t)32 * NB + 32, 0, 1);
        #pragma unroll
        for (int mt = 0; mt < 2; ++mt)
            #pragma unroll
            for (int nt = 0; nt < 4; ++nt)
                acc[mt][nt] = __builtin_amdgcn_wmma_f32_16x16x32_f16(
                    /*neg_a=*/false, a[mt], /*neg_b=*/false, b[nt],
                    /*c_mod=*/(short)0, acc[mt][nt],
                    /*reuse_a=*/false, /*reuse_b=*/false);
        ap += 32;
        bp += (size_t)32 * NB;
    }

    // Epilogue: C/D layout — VGPR r: M = r + 8*(lane>=16), N = lane & 15.
    #pragma unroll
    for (int nt = 0; nt < 4; ++nt) {
        const int j = col0 + nt * 16 + (lane & 15);
        const float nyj = ny[j];
        const float* eyp = ey + (size_t)j * 4;
        const float ey0 = eyp[0], ey1 = eyp[1], ey2 = eyp[2], ey3 = eyp[3];
        #pragma unroll
        for (int mt = 0; mt < 2; ++mt) {
            const int mbase = row0 + mt * 16 + (lane >> 4) * 8;
            #pragma unroll
            for (int r = 0; r < 8; ++r) {
                int i = mbase + r;
                float sq  = nx[i] + nyj - 2.0f * acc[mt][nt][r];
                float raw = __expf(-fmaxf(sq, 0.f));             // GAMMA = 1.0
                const float* exp_ = ex + (size_t)i * 4;
                float q = __cosf((exp_[0] - ey0) * 0.5f);
                q *= __cosf((exp_[1] - ey1) * 0.5f);
                q *= __cosf((exp_[2] - ey2) * 0.5f);
                q *= __cosf((exp_[3] - ey3) * 0.5f);
                out[(size_t)i * NB + j] = 0.5f * raw + 0.5f * q;
            }
        }
    }
}

// ---------------------------------------------------------------------------
extern "C" void kernel_launch(void* const* d_in, const int* in_sizes, int n_in,
                              void* d_out, int out_size, void* d_ws, size_t ws_size,
                              hipStream_t stream)
{
    (void)in_sizes; (void)n_in; (void)out_size; (void)ws_size;
    const float* x   = (const float*)d_in[0];
    const float* y   = (const float*)d_in[1];
    const float* c1w = (const float*)d_in[2];
    const float* c1b = (const float*)d_in[3];
    const float* c2w = (const float*)d_in[4];
    const float* c2b = (const float*)d_in[5];
    const float* f1w = (const float*)d_in[6];
    const float* f1b = (const float*)d_in[7];
    const float* f2w = (const float*)d_in[8];
    const float* f2b = (const float*)d_in[9];
    const float* bng = (const float*)d_in[10];
    const float* bnb = (const float*)d_in[11];
    float* out = (float*)d_out;

    // Workspace layout (~13.4 MB total)
    char* ws = (char*)d_ws;
    const size_t XH_BYTES = (size_t)NB * KPAD * sizeof(_Float16);  // 6,553,600
    _Float16* xh  = (_Float16*)(ws);
    _Float16* yhT = (_Float16*)(ws + XH_BYTES);
    float* nx  = (float*)(ws + 2 * XH_BYTES);
    float* ny  = nx + NB;
    float* eAr = ny + NB;        // raw embeddings, x batch (4096x4)
    float* eBr = eAr + NB * 4;   // raw embeddings, y batch
    float* eA  = eBr + NB * 4;   // BN-normalized
    float* eB  = eA  + NB * 4;

    // 1) f16 conversion + row norms (x row-major, y transposed for B-fragments)
    convert_rows_kernel<<<NB, 256, 0, stream>>>(x, xh, nx);
    convert_cols_kernel<<<NB, 256, 0, stream>>>(y, yhT, ny);

    // 2) CNN embeddings (pre-BN), one block per image
    embed_kernel<<<NB, 256, 0, stream>>>(x, c1w, c1b, c2w, c2b, f1w, f1b, f2w, f2b, eAr);
    embed_kernel<<<NB, 256, 0, stream>>>(y, c1w, c1b, c2w, c2b, f1w, f1b, f2w, f2b, eBr);

    // 3) training-mode BatchNorm with each batch's own stats
    bn_kernel<<<1, 256, 0, stream>>>(eAr, bng, bnb, eA);
    bn_kernel<<<1, 256, 0, stream>>>(eBr, bng, bnb, eB);

    // 4) fused WMMA GEMM + RBF/quantum epilogue: 128x128 block tiles, grid 32x32
    dim3 grid(NB / 128, NB / 128);
    fused_rbf_wmma_kernel<<<grid, 256, 0, stream>>>(xh, yhT, nx, ny, eA, eB, out);
}
